// MaskedSelfAttention_71691594104857
// MI455X (gfx1250) — matrix operand order repaired and checked by reading, not compiled, not run on hardware
//
#include <hip/hip_runtime.h>
#include <hip/hip_bf16.h>

typedef __attribute__((ext_vector_type(16))) _Float16 v16h;
typedef __attribute__((ext_vector_type(8)))  _Float16 v8h;
typedef __attribute__((ext_vector_type(8)))  float    v8f;

constexpr int EMB    = 512;
constexpr int DHEAD  = 64;
constexpr int NBATCH = 4;
constexpr int NSEQ   = 4096;
constexpr int MROWS  = NBATCH * NSEQ;     // 16384
constexpr int NTILES = MROWS / 16;        // 1024 query tiles
constexpr int CHUNK  = 1024;              // keys per split-K chunk
constexpr int MAXCH  = NSEQ / CHUNK;      // 4 chunks per q-tile (max)
constexpr int PSTR   = 16 + 16 + 16 * DHEAD;  // floats per partial: m,l,O

static __device__ __forceinline__ v8f wmma16(v16h a, v16h b, v8f c) {
  // D = A(16x32 f16) * B(32x16 f16) + C(16x16 f32)
  return __builtin_amdgcn_wmma_f32_16x16x32_f16(false, a, false, b, (short)0, c,
                                                false, false);
}

// ---------------------------------------------------------------------------
// Kernel 1: fused QKV projection.  X[16384,512] x W[512,64] (+bias) -> f16.
// Q is pre-scaled by log2(e)/sqrt(64) so the attention softmax runs in base-2.
// V is stored transposed: Vt[b][64][4096].
// One wave per 16x16 output tile; tiles = 1024(m) * 4(col) * 3(mat) = 12288.
// ---------------------------------------------------------------------------
__global__ __launch_bounds__(256) void qkv_kernel(
    const float* __restrict__ x,
    const float* __restrict__ Wq, const float* __restrict__ bq,
    const float* __restrict__ Wk, const float* __restrict__ bk,
    const float* __restrict__ Wv, const float* __restrict__ bv,
    _Float16* __restrict__ Qh, _Float16* __restrict__ Kh,
    _Float16* __restrict__ Vt)
{
  const int lane = threadIdx.x & 31;
  const int wave = threadIdx.x >> 5;
  const int wid  = blockIdx.x * 8 + wave;
  const int mat  = wid % 3;          // 0=Q 1=K 2=V
  const int ct   = (wid / 3) & 3;    // column tile (64/16)
  const int mt   = wid / 12;         // row tile (16384/16)
  const int half = lane >> 4;
  const int lm   = lane & 15;
  const int m0   = mt * 16;
  const int col  = ct * 16 + lm;

  const float* W; const float* bias;
  if (mat == 0)      { W = Wq; bias = bq; }
  else if (mat == 1) { W = Wk; bias = bk; }
  else               { W = Wv; bias = bv; }

  const float* xrow = x + (size_t)(m0 + lm) * EMB;
  v8f acc = {};

  for (int kb = 0; kb < EMB; kb += 32) {
    // A operand: rows of x, k = kb + half*16 + {0..15}
    v16h a;
    const float* p0 = xrow + kb + half * 16;
    #pragma unroll
    for (int j = 0; j < 8; ++j) {
      a[j]     = (_Float16)p0[j];
      a[j + 8] = (_Float16)p0[j + 8];
    }
    // B operand: W[k][col], k = kb + half*16 + j   (stride-DHEAD gather)
    v16h bop;
    const float* wp = W + (size_t)(kb + half * 16) * DHEAD + col;
    #pragma unroll
    for (int j = 0; j < 16; ++j) bop[j] = (_Float16)wp[(size_t)j * DHEAD];

    acc = wmma16(a, bop, acc);
  }

  const float bsc = bias[col];
  #pragma unroll
  for (int i = 0; i < 8; ++i) acc[i] += bsc;

  if (mat == 0) {
    // fold 1/sqrt(64) and log2(e) into Q: softmax computed with exp2
    constexpr float QSCALE = 0.125f * 1.44269504088896340736f;
    #pragma unroll
    for (int i = 0; i < 8; ++i) acc[i] *= QSCALE;
  }

  if (mat == 2) {
    #pragma unroll
    for (int i = 0; i < 8; ++i) {
      const int r = m0 + half * 8 + i;
      const int b = r >> 12;
      const int n = r & 4095;
      Vt[(size_t)(b * DHEAD + col) * NSEQ + n] = (_Float16)acc[i];
    }
  } else {
    _Float16* dst = (mat == 0) ? Qh : Kh;
    #pragma unroll
    for (int i = 0; i < 8; ++i) {
      const int r = m0 + half * 8 + i;
      dst[(size_t)r * DHEAD + col] = (_Float16)acc[i];
    }
  }
}

// ---------------------------------------------------------------------------
// Kernel 2: split-K flash attention (causal, online softmax, base-2).
// One wave per (16-query tile, 1024-key chunk): up to 4096 units.
// Per 32-key step: 4 WMMA for S(16x32), LDS transpose of P, 4 WMMA for P*V.
// Writes unnormalized partials (m, l, O) for the combine kernel.
// ---------------------------------------------------------------------------
__global__ __launch_bounds__(256) void attn_kernel(
    const _Float16* __restrict__ Qh, const _Float16* __restrict__ Kh,
    const _Float16* __restrict__ Vt, float* __restrict__ Part)
{
  __shared__ __align__(16) _Float16 lds[8][16][32];   // 1KB per wave

  const int lane  = threadIdx.x & 31;
  const int wave  = threadIdx.x >> 5;
  const int wid   = blockIdx.x * 8 + wave;    // unit id
  const int chunk = wid & (MAXCH - 1);
  const int tile  = wid >> 2;                 // b*256 + qt
  const int b     = tile >> 8;
  const int qt    = tile & 255;
  const int q0    = qt * 16;
  const int half  = lane >> 4;
  const int lm    = lane & 15;
  const size_t brow = (size_t)b * NSEQ;

  const int nkb     = (q0 + 47) >> 5;               // causal 32-key blocks
  const int kbStart = chunk << 5;                   // CHUNK/32 = 32
  const int kbEnd   = min(nkb, (chunk + 1) << 5);
  if (kbStart >= kbEnd) return;                     // chunk beyond causal edge

  // Q A-operands (fixed for the whole key loop): d 0..31 and 32..63
  v16h aq0, aq1;
  {
    const _Float16* qrow = Qh + (brow + q0 + lm) * DHEAD;
    const v8h q00 = *(const v8h*)(qrow + half * 16);
    const v8h q01 = *(const v8h*)(qrow + half * 16 + 8);
    const v8h q10 = *(const v8h*)(qrow + 32 + half * 16);
    const v8h q11 = *(const v8h*)(qrow + 40 + half * 16);
    #pragma unroll
    for (int j = 0; j < 8; ++j) {
      aq0[j] = q00[j]; aq0[j + 8] = q01[j];
      aq1[j] = q10[j]; aq1[j + 8] = q11[j];
    }
  }

  v8f O0 = {}, O1 = {}, O2 = {}, O3 = {};
  float M8[8], L8[8];
  #pragma unroll
  for (int i = 0; i < 8; ++i) { M8[i] = -3.0e38f; L8[i] = 0.0f; }

  for (int kb = kbStart; kb < kbEnd; ++kb) {
    const int key0 = kb << 5;

    const _Float16* krow0 = Kh + (brow + key0 + lm) * DHEAD;  // S0 cols
    const _Float16* krow1 = krow0 + 16 * DHEAD;               // S1 cols
    const v16h bk00 = *(const v16h*)(krow0 + half * 16);
    const v16h bk01 = *(const v16h*)(krow0 + 32 + half * 16);
    const v16h bk10 = *(const v16h*)(krow1 + half * 16);
    const v16h bk11 = *(const v16h*)(krow1 + 32 + half * 16);

    if (kb + 1 < kbEnd) __builtin_prefetch(krow0 + 32 * DHEAD, 0, 0);

    v8f S0 = {}, S1 = {};
    S0 = wmma16(aq0, bk00, S0);
    S0 = wmma16(aq1, bk01, S0);
    S1 = wmma16(aq0, bk10, S1);
    S1 = wmma16(aq1, bk11, S1);

    if (key0 + 31 > q0) {                          // diagonal block: mask
      const int keyA = key0 + lm;
      #pragma unroll
      for (int i = 0; i < 8; ++i) {
        const int qr = q0 + half * 8 + i;
        if (keyA > qr)       S0[i] = -3.0e38f;
        if (keyA + 16 > qr)  S1[i] = -3.0e38f;
      }
    }

    // row max across the 16 lanes of each half
    float rmax[8], rsum[8];
    #pragma unroll
    for (int i = 0; i < 8; ++i) rmax[i] = fmaxf(S0[i], S1[i]);
    #pragma unroll
    for (int off = 1; off < 16; off <<= 1) {
      #pragma unroll
      for (int i = 0; i < 8; ++i)
        rmax[i] = fmaxf(rmax[i], __shfl_xor(rmax[i], off, 32));
    }

    float P0[8], P1[8];
    #pragma unroll
    for (int i = 0; i < 8; ++i) {
      const float nm = fmaxf(M8[i], rmax[i]);
      const float cs = exp2f(M8[i] - nm);          // v_exp_f32 native base-2
      P0[i] = exp2f(S0[i] - nm);
      P1[i] = exp2f(S1[i] - nm);
      rsum[i] = P0[i] + P1[i];
      M8[i] = nm;
      L8[i] *= cs;
      O0[i] *= cs; O1[i] *= cs; O2[i] *= cs; O3[i] *= cs;
    }
    #pragma unroll
    for (int off = 1; off < 16; off <<= 1) {
      #pragma unroll
      for (int i = 0; i < 8; ++i)
        rsum[i] += __shfl_xor(rsum[i], off, 32);
    }
    #pragma unroll
    for (int i = 0; i < 8; ++i) L8[i] += rsum[i];

    // C-layout -> A-layout transpose of P through LDS
    #pragma unroll
    for (int i = 0; i < 8; ++i) {
      lds[wave][half * 8 + i][lm]      = (_Float16)P0[i];
      lds[wave][half * 8 + i][lm + 16] = (_Float16)P1[i];
    }
    asm volatile("s_wait_dscnt 0x0" ::: "memory");
    v16h aP;
    {
      const v8h p0 = *(const v8h*)&lds[wave][lm][half * 16];
      const v8h p1 = *(const v8h*)&lds[wave][lm][half * 16 + 8];
      #pragma unroll
      for (int j = 0; j < 8; ++j) { aP[j] = p0[j]; aP[j + 8] = p1[j]; }
    }

    // V B-operands (Vt is [b][64][4096], contiguous in key)
    const _Float16* vbase =
        Vt + (size_t)(b * DHEAD + lm) * NSEQ + key0 + half * 16;
    const v16h bv0 = *(const v16h*)(vbase);
    const v16h bv1 = *(const v16h*)(vbase + 16 * NSEQ);
    const v16h bv2 = *(const v16h*)(vbase + 32 * NSEQ);
    const v16h bv3 = *(const v16h*)(vbase + 48 * NSEQ);
    O0 = wmma16(aP, bv0, O0);
    O1 = wmma16(aP, bv1, O1);
    O2 = wmma16(aP, bv2, O2);
    O3 = wmma16(aP, bv3, O3);
  }

  // write unnormalized partial: [m(16) | l(16) | O(16x64)]
  float* pbase = Part + (size_t)wid * PSTR;
  if (lm == 0) {
    #pragma unroll
    for (int i = 0; i < 8; ++i) {
      pbase[half * 8 + i]      = M8[i];
      pbase[16 + half * 8 + i] = L8[i];
    }
  }
  #pragma unroll
  for (int i = 0; i < 8; ++i) {
    float* po = pbase + 32 + (size_t)(half * 8 + i) * DHEAD;
    po[lm]      = O0[i];
    po[16 + lm] = O1[i];
    po[32 + lm] = O2[i];
    po[48 + lm] = O3[i];
  }
}

// ---------------------------------------------------------------------------
// Kernel 2b: combine split-K partials (log-sum-exp merge, base-2) -> Oh f16.
// One thread per output element: 1024 tiles * (16*64) elements.
// ---------------------------------------------------------------------------
__global__ __launch_bounds__(256) void combine_kernel(
    const float* __restrict__ Part, _Float16* __restrict__ Oh)
{
  const int gid  = blockIdx.x * 256 + threadIdx.x;
  const int tile = gid >> 10;          // b*256 + qt
  const int e    = gid & 1023;
  const int r    = e >> 6;
  const int c    = e & 63;
  const int q0   = (tile & 255) * 16;
  const int nch  = (q0 + 16 + CHUNK - 1) >> 10;

  const float* base = Part + (size_t)tile * MAXCH * PSTR;

  float M = -3.0e38f;
  for (int ch = 0; ch < nch; ++ch)
    M = fmaxf(M, base[(size_t)ch * PSTR + r]);

  float l = 0.0f, o = 0.0f;
  for (int ch = 0; ch < nch; ++ch) {
    const float* p = base + (size_t)ch * PSTR;
    const float w = exp2f(p[r] - M);
    l += w * p[16 + r];
    o += w * p[32 + (size_t)r * DHEAD + c];
  }

  Oh[(size_t)(tile * 16 + r) * DHEAD + c] = (_Float16)(o / l);
}

// ---------------------------------------------------------------------------
// Kernel 3: output projection.  O[16384,64] x Wo[64,512] + bo -> f32 out.
// One wave per 16x16 tile: 1024(m) * 32(col) = 32768 waves.
// ---------------------------------------------------------------------------
__global__ __launch_bounds__(256) void outproj_kernel(
    const _Float16* __restrict__ Oh, const float* __restrict__ Wo,
    const float* __restrict__ bo, float* __restrict__ out)
{
  const int lane = threadIdx.x & 31;
  const int wave = threadIdx.x >> 5;
  const int wid  = blockIdx.x * 8 + wave;
  const int ct   = wid & 31;
  const int mt   = wid >> 5;
  const int m0   = mt * 16;
  const int half = lane >> 4;
  const int lm   = lane & 15;
  const int col  = ct * 16 + lm;

  v16h a0, a1;
  {
    const _Float16* orow = Oh + (size_t)(m0 + lm) * DHEAD;
    const v8h t00 = *(const v8h*)(orow + half * 16);
    const v8h t01 = *(const v8h*)(orow + half * 16 + 8);
    const v8h t10 = *(const v8h*)(orow + 32 + half * 16);
    const v8h t11 = *(const v8h*)(orow + 40 + half * 16);
    #pragma unroll
    for (int j = 0; j < 8; ++j) {
      a0[j] = t00[j]; a0[j + 8] = t01[j];
      a1[j] = t10[j]; a1[j + 8] = t11[j];
    }
  }

  v16h b0, b1;
  {
    const float* wp = Wo + (size_t)(half * 16) * EMB + col;
    #pragma unroll
    for (int j = 0; j < 16; ++j) {
      b0[j] = (_Float16)wp[(size_t)j * EMB];          // k = half*16 + j
      b1[j] = (_Float16)wp[(size_t)(j + 32) * EMB];   // k = 32 + half*16 + j
    }
  }

  v8f acc = {};
  acc = wmma16(a0, b0, acc);
  acc = wmma16(a1, b1, acc);

  const float bsc = bo[col];
  #pragma unroll
  for (int i = 0; i < 8; ++i)
    out[(size_t)(m0 + half * 8 + i) * EMB + col] = acc[i] + bsc;
}

// ---------------------------------------------------------------------------
extern "C" void kernel_launch(void* const* d_in, const int* in_sizes, int n_in,
                              void* d_out, int out_size, void* d_ws,
                              size_t ws_size, hipStream_t stream) {
  const float* x  = (const float*)d_in[0];
  const float* Wq = (const float*)d_in[1];
  const float* bq = (const float*)d_in[2];
  const float* Wk = (const float*)d_in[3];
  const float* bk = (const float*)d_in[4];
  const float* Wv = (const float*)d_in[5];
  const float* bv = (const float*)d_in[6];
  const float* Wo = (const float*)d_in[7];
  const float* bo = (const float*)d_in[8];
  float* out = (float*)d_out;

  _Float16* qh = (_Float16*)d_ws;                       // 2 MB each
  _Float16* kh = qh + (size_t)MROWS * DHEAD;
  _Float16* vt = kh + (size_t)MROWS * DHEAD;
  _Float16* oh = vt + (size_t)MROWS * DHEAD;
  float* part  = (float*)(oh + (size_t)MROWS * DHEAD);  // ~17.3 MB partials

  qkv_kernel<<<1536, 256, 0, stream>>>(x, Wq, bq, Wk, bk, Wv, bv, qh, kh, vt);
  attn_kernel<<<(NTILES * MAXCH) / 8, 256, 0, stream>>>(qh, kh, vt, part);
  combine_kernel<<<(NTILES * 16 * DHEAD) / 256, 256, 0, stream>>>(part, oh);
  outproj_kernel<<<(NTILES * (EMB / 16)) / 8, 256, 0, stream>>>(oh, Wo, bo, out);
}
